// LSTMDecoder_9259949490783
// MI455X (gfx1250) — matrix operand order repaired and checked by reading, not compiled, not run on hardware
//
#include <hip/hip_runtime.h>

// ---------------------------------------------------------------------------
// LSTM decoder (B=64, H=1024, L=2, T=128) for gfx1250 (MI455X).
// Strategy: convert weights fp32->bf16 once per launch, then 128 serial steps
// of {layer0 fused GEMM+cell, layer1 fused GEMM+cell, FC} using
// v_wmma_f32_16x16x32_bf16 with fp32 accumulators. Weights (34 MB bf16) live
// in the 192 MB L2 across all 128 steps; WGP-scope prefetch runs the
// L2 -> WGP$ refill ahead of the load clauses.
// ---------------------------------------------------------------------------

typedef __bf16 bf16;
typedef __attribute__((ext_vector_type(16))) __bf16 v16bf;
typedef __attribute__((ext_vector_type(8)))  __bf16 v8bf;
typedef __attribute__((ext_vector_type(8)))  float  v8f;

#define LSTM_H  1024
#define LSTM_B  64
#define LSTM_L  2
#define LSTM_T  128
#define LSTM_BH (LSTM_B * LSTM_H)          // 65536
#define LSTM_HH ((size_t)LSTM_H * LSTM_H)  // 1048576 elements per gate block

__device__ __forceinline__ v16bf cat8(v8bf lo, v8bf hi) {
  return __builtin_shufflevector(lo, hi, 0, 1, 2, 3, 4, 5, 6, 7,
                                         8, 9, 10, 11, 12, 13, 14, 15);
}

__device__ __forceinline__ v8f wmma_bf16(v16bf a, v16bf b, v8f c) {
  // (neg_a, A, neg_b, B, c_mod, C, reuse_a, reuse_b)
  return __builtin_amdgcn_wmma_f32_16x16x32_bf16(false, a, false, b,
                                                 (short)0, c, false, false);
}

__device__ __forceinline__ float fsigmoid(float x) {
  return 1.0f / (1.0f + __expf(-x));
}
__device__ __forceinline__ float ftanh(float x) {
  float e2 = __expf(2.0f * x);
  return (e2 - 1.0f) / (e2 + 1.0f);
}

// A-tile (16x32 bf16): lanes 0-15 -> row m0+lane, K = k0+0..15
//                      lanes 16-31 -> row m0+lane-16, K = k0+16..31
__device__ __forceinline__ v16bf load_a(const bf16* __restrict__ row, int k0) {
  return cat8(*(const v8bf*)(row + k0), *(const v8bf*)(row + k0 + 8));
}
// B-tile (32x16 bf16): lane L(0-15) -> col n0+L (= W row, contiguous K),
//                      K = k0+0..15; lane L+16 -> same col, K = k0+16..31
__device__ __forceinline__ v16bf load_b(const bf16* __restrict__ row, int k0) {
  return cat8(*(const v8bf*)(row + k0), *(const v8bf*)(row + k0 + 8));
}

// ------------------------- fp32 -> bf16 weight convert ----------------------
__global__ __launch_bounds__(256) void cvt_f32_bf16(const float* __restrict__ in,
                                                    bf16* __restrict__ out, int n) {
  int i = blockIdx.x * 256 + threadIdx.x;
  if (i < n) out[i] = (bf16)in[i];
}

// ------------------------- state init (per launch) --------------------------
__global__ __launch_bounds__(256) void init_state(const float* __restrict__ hin,
                                                  bf16* __restrict__ xb0,
                                                  bf16* __restrict__ h0b0,
                                                  bf16* __restrict__ h1b0,
                                                  float* __restrict__ c) {
  int i = blockIdx.x * 256 + threadIdx.x;     // grid covers BH exactly
  xb0[i]  = (bf16)hin[i];
  h0b0[i] = (bf16)0.0f;
  h1b0[i] = (bf16)0.0f;
  c[i]           = 0.0f;
  c[i + LSTM_BH] = 0.0f;
}

// ---------------- fused LSTM layer step: gates GEMM + cell ------------------
// Each wave: one 16(batch) x 16(unit) tile, all 4 gates accumulated in regs.
// 256 wave-tiles total -> 32 blocks x 8 waves.
__global__ __launch_bounds__(256) void lstm_layer_step(
    const bf16* __restrict__ x,      // [B,H] layer input (bf16)
    const bf16* __restrict__ hprev,  // [B,H] this layer's previous h (bf16)
    const bf16* __restrict__ Wih,    // [4H,H] bf16, gate order i,f,g,o
    const bf16* __restrict__ Whh,    // [4H,H] bf16
    const float* __restrict__ bih,   // [4H]
    const float* __restrict__ bhh,   // [4H]
    float* __restrict__ cst,         // [B,H] cell state (in/out, fp32)
    bf16* __restrict__ hout,         // [B,H] new h (bf16)
    bf16* __restrict__ hout2)        // optional second copy (feedback buffer)
{
  const int lane = threadIdx.x & 31;
  const int tile = blockIdx.x * 8 + (threadIdx.x >> 5);  // 0..255
  const int m0 = (tile & 3) << 4;                        // batch tile
  const int n0 = (tile >> 2) << 4;                       // hidden-unit tile

  v8f acc0 = {}, acc1 = {}, acc2 = {}, acc3 = {};

  const int am = m0 + (lane & 15);
  const int ak = (lane & 16) ? 16 : 0;
  const bf16* __restrict__ xrow = x + (size_t)am * LSTM_H + ak;
  const bf16* __restrict__ hrow = hprev + (size_t)am * LSTM_H + ak;

  const int bn = n0 + (lane & 15);
  const int bk = (lane & 16) ? 16 : 0;
  const bf16* __restrict__ w0 = Wih + (size_t)bn * LSTM_H + bk;  // gate i rows
  const bf16* __restrict__ u0 = Whh + (size_t)bn * LSTM_H + bk;

  for (int k0 = 0; k0 < LSTM_H; k0 += 32) {
    // WGP-scope prefetch of the weight streams two K-iterations (128 B) ahead:
    // pulls L2-resident lines into the WGP cache before the load clauses.
    __builtin_prefetch(w0 + k0 + 64, 0, 3);
    __builtin_prefetch(u0 + k0 + 64, 0, 3);

    v16bf ax = load_a(xrow, k0);
    v16bf ah = load_a(hrow, k0);

    v16bf b;
    b = load_b(w0, k0);                     acc0 = wmma_bf16(ax, b, acc0);
    b = load_b(u0, k0);                     acc0 = wmma_bf16(ah, b, acc0);
    b = load_b(w0 + LSTM_HH, k0);           acc1 = wmma_bf16(ax, b, acc1);
    b = load_b(u0 + LSTM_HH, k0);           acc1 = wmma_bf16(ah, b, acc1);
    b = load_b(w0 + 2 * LSTM_HH, k0);       acc2 = wmma_bf16(ax, b, acc2);
    b = load_b(u0 + 2 * LSTM_HH, k0);       acc2 = wmma_bf16(ah, b, acc2);
    b = load_b(w0 + 3 * LSTM_HH, k0);       acc3 = wmma_bf16(ax, b, acc3);
    b = load_b(u0 + 3 * LSTM_HH, k0);       acc3 = wmma_bf16(ah, b, acc3);
  }

  // biases: depend only on this lane's unit index n
  const int n = n0 + (lane & 15);
  const float bi = bih[0 * LSTM_H + n] + bhh[0 * LSTM_H + n];
  const float bf = bih[1 * LSTM_H + n] + bhh[1 * LSTM_H + n];
  const float bg = bih[2 * LSTM_H + n] + bhh[2 * LSTM_H + n];
  const float bo = bih[3 * LSTM_H + n] + bhh[3 * LSTM_H + n];

  // C/D layout: VGPR r, lanes 0-15 -> (m0+r, n); lanes 16-31 -> (m0+8+r, n)
  const int mb = m0 + ((lane & 16) ? 8 : 0);
#pragma unroll
  for (int r = 0; r < 8; ++r) {
    const size_t idx = (size_t)(mb + r) * LSTM_H + n;
    float gi = acc0[r] + bi;
    float gf = acc1[r] + bf;
    float gg = acc2[r] + bg;
    float go = acc3[r] + bo;
    float cp = cst[idx];
    float cn = fsigmoid(gf) * cp + fsigmoid(gi) * ftanh(gg);
    float hn = fsigmoid(go) * ftanh(cn);
    cst[idx] = cn;
    bf16 hb = (bf16)hn;
    hout[idx] = hb;
    if (hout2) hout2[idx] = hb;
  }
}

// --------------------------- FC projection step -----------------------------
__global__ __launch_bounds__(256) void fc_step(const bf16* __restrict__ h1,
                                               const bf16* __restrict__ Wfc,
                                               const float* __restrict__ bfc,
                                               float* __restrict__ out) {
  const int lane = threadIdx.x & 31;
  const int tile = blockIdx.x * 8 + (threadIdx.x >> 5);  // 0..255
  const int m0 = (tile & 3) << 4;
  const int n0 = (tile >> 2) << 4;

  v8f acc = {};
  const bf16* __restrict__ arow =
      h1 + (size_t)(m0 + (lane & 15)) * LSTM_H + ((lane & 16) ? 16 : 0);
  const bf16* __restrict__ wrow =
      Wfc + (size_t)(n0 + (lane & 15)) * LSTM_H + ((lane & 16) ? 16 : 0);

  for (int k0 = 0; k0 < LSTM_H; k0 += 32) {
    __builtin_prefetch(wrow + k0 + 64, 0, 3);
    v16bf a = load_a(arow, k0);
    v16bf b = load_b(wrow, k0);
    acc = wmma_bf16(a, b, acc);
  }

  const int n = n0 + (lane & 15);
  const float bb = bfc[n];
  const int mb = m0 + ((lane & 16) ? 8 : 0);
#pragma unroll
  for (int r = 0; r < 8; ++r) {
    out[(size_t)(mb + r) * LSTM_H + n] = acc[r] + bb;
  }
}

// ----------------------------------------------------------------------------
extern "C" void kernel_launch(void* const* d_in, const int* in_sizes, int n_in,
                              void* d_out, int out_size, void* d_ws, size_t ws_size,
                              hipStream_t stream) {
  const float* h    = (const float*)d_in[0];  // [B,H]
  const float* Wih  = (const float*)d_in[1];  // [L,4H,H]
  const float* Whh  = (const float*)d_in[2];  // [L,4H,H]
  const float* bih  = (const float*)d_in[3];  // [L,4H]
  const float* bhh  = (const float*)d_in[4];  // [L,4H]
  const float* Wfc  = (const float*)d_in[5];  // [H,H]
  const float* bfc  = (const float*)d_in[6];  // [H]
  float* out = (float*)d_out;                 // [T,B,H]

  char* ws = (char*)d_ws;
  const size_t MB = 1024 * 1024;
  bf16* WihB = (bf16*)(ws);                 // 2*4H*H = 8M elems = 16 MB
  bf16* WhhB = (bf16*)(ws + 16 * MB);       // 16 MB
  bf16* WfcB = (bf16*)(ws + 32 * MB);       // 1M elems = 2 MB
  bf16* xb   = (bf16*)(ws + 34 * MB);       // 2*[B,H] bf16 ping-pong (feedback)
  bf16* h0b  = (bf16*)(ws + 34 * MB + 256 * 1024);
  bf16* h1b  = (bf16*)(ws + 34 * MB + 512 * 1024);
  float* cst = (float*)(ws + 34 * MB + 768 * 1024);  // 2*[B,H] fp32 = 512 KB

  const int nW = LSTM_L * 4 * LSTM_H * LSTM_H;  // 8388608
  const int nF = LSTM_H * LSTM_H;               // 1048576

  cvt_f32_bf16<<<nW / 256, 256, 0, stream>>>(Wih, WihB, nW);
  cvt_f32_bf16<<<nW / 256, 256, 0, stream>>>(Whh, WhhB, nW);
  cvt_f32_bf16<<<nF / 256, 256, 0, stream>>>(Wfc, WfcB, nF);

  // t=0 reads xb[0]=bf16(h), h0b[0]=0, h1b[0]=0, c=0
  init_state<<<LSTM_BH / 256, 256, 0, stream>>>(h, xb, h0b, h1b, cst);

  const size_t LOFF = (size_t)4 * LSTM_H * LSTM_H;  // per-layer weight offset
  for (int t = 0; t < LSTM_T; ++t) {
    const int pp = t & 1;       // buffers produced at step t-1
    const int np = pp ^ 1;      // buffers produced at step t
    // layer 0: x = fed-back h1[t-1] (or initial h), own state h0
    lstm_layer_step<<<32, 256, 0, stream>>>(
        xb + (size_t)pp * LSTM_BH, h0b + (size_t)pp * LSTM_BH,
        WihB, WhhB, bih, bhh,
        cst, h0b + (size_t)np * LSTM_BH, (bf16*)nullptr);
    // layer 1: x = h0[t], own state h1; also writes feedback copy into xb
    lstm_layer_step<<<32, 256, 0, stream>>>(
        h0b + (size_t)np * LSTM_BH, h1b + (size_t)pp * LSTM_BH,
        WihB + LOFF, WhhB + LOFF, bih + 4 * LSTM_H, bhh + 4 * LSTM_H,
        cst + LSTM_BH, h1b + (size_t)np * LSTM_BH, xb + (size_t)np * LSTM_BH);
    // out[t] = h1[t] @ Wfc^T + bfc
    fc_step<<<32, 256, 0, stream>>>(
        h1b + (size_t)np * LSTM_BH, WfcB, bfc, out + (size_t)t * LSTM_BH);
  }
}
